// CustomLSTM_833223655490
// MI455X (gfx1250) — matrix operand order repaired and checked by reading, not compiled, not run on hardware
//
#include <hip/hip_runtime.h>

#define SEQ   512
#define BS    128
#define IN_SZ 256
#define HS    512
#define KTOT  (IN_SZ + HS)     // 768
#define FOURH (4 * HS)         // 2048
#define NKC   (KTOT / 32)      // 24 K-chunks of 32
#define CHUNK (32 * 16)        // elements per (col-block, k-chunk) fragment group

typedef __bf16       v16bf __attribute__((ext_vector_type(16)));
typedef float        v8f   __attribute__((ext_vector_type(8)));
typedef unsigned int v4u   __attribute__((ext_vector_type(4)));
typedef unsigned int v2u   __attribute__((ext_vector_type(2)));
typedef float        v4f   __attribute__((ext_vector_type(4)));

union Frag16 { v16bf bf; v4u u4[2]; };
union Pack4  { __bf16 b[4]; v2u u; };

// Pre-swizzle [W;U] (f32) into bf16 WMMA-B fragment order:
// Vswz[cb][kc][lane][elem], cb = column block of 16, kc = K-chunk of 32.
// B layout used here (contiguous k, same as A): lane = col + 16*(krow/16), elem = krow%16.
__global__ void build_weights(const float* __restrict__ W,
                              const float* __restrict__ U,
                              __bf16* __restrict__ Vswz) {
  int i = blockIdx.x * blockDim.x + threadIdx.x;
  if (i >= KTOT * FOURH) return;
  int k = i / FOURH, n = i % FOURH;
  float v = (k < IN_SZ) ? W[i] : U[i - IN_SZ * FOURH];
  int cb = n >> 4, cc = n & 15;
  int kc = k >> 5, r = k & 31;
  int lane = cc + ((r >> 4) << 4);
  int elem = r & 15;
  Vswz[(((size_t)cb * NKC + kc) * 32 + lane) * 16 + elem] = (__bf16)v;
}

// One-time f32 -> bf16 conversion of the whole x tensor (4 elems / thread).
__global__ void convert_x(const float* __restrict__ x,
                          __bf16* __restrict__ xb) {
  size_t i = (size_t)blockIdx.x * blockDim.x + threadIdx.x;
  if (i >= (size_t)SEQ * BS * IN_SZ / 4) return;
  v4f v = *(const v4f*)(x + 4 * i);
  Pack4 p;
  #pragma unroll
  for (int e = 0; e < 4; ++e) p.b[e] = (__bf16)v[e];
  *(v2u*)(xb + 4 * i) = p.u;
}

// Zero initial h (bf16) and c (f32); precompute bias_w + bias_u.
__global__ void init_state(float* __restrict__ c, __bf16* __restrict__ h0,
                           const float* __restrict__ bw,
                           const float* __restrict__ bu,
                           float* __restrict__ bsum) {
  int i = blockIdx.x * blockDim.x + threadIdx.x;
  if (i < BS * HS) { c[i] = 0.f; h0[i] = (__bf16)0.f; }
  if (i < FOURH)   { bsum[i] = bw[i] + bu[i]; }
}

// One LSTM timestep. 128 threads = 4 waves; wave g computes gate g's 16x16
// f32 tile of gates = [x_t | h] @ [W;U] for rows [m0,m0+16), hidden cols
// [n0,n0+16), then the block completes the cell update locally.
__launch_bounds__(128)
__global__ void lstm_step(const __bf16* __restrict__ xb_t,     // [BS, IN_SZ] bf16
                          const __bf16* __restrict__ Vswz,     // swizzled weights
                          const float*  __restrict__ bsum,     // [4H] bias_w+bias_u
                          const __bf16* __restrict__ h_in,     // [BS, HS] bf16
                          __bf16*       __restrict__ h_out_bf, // [BS, HS] bf16
                          float*        __restrict__ c_state,  // [BS, HS] f32
                          float*        __restrict__ h_out_f32)// [BS, HS] f32
{
  __shared__ float Glds[4][16][16];     // activated gate tiles (i,f,g,o)

  const int tid  = threadIdx.x;
  const int wave = tid >> 5;            // gate id: 0=i 1=f 2=g 3=o
  const int lane = tid & 31;
  const int half = lane >> 4;
  const int lr   = lane & 15;

  const int m0 = blockIdx.y * 16;       // batch rows
  const int n0 = blockIdx.x * 16;       // hidden cols
  const int cb = wave * (HS / 16) + blockIdx.x;   // col block in [0, 4H/16)

  // Per-lane fragment base pointers (all 16B-aligned).
  // A-layout (matches B swizzle): per lane, one contiguous 16-elem K
  // run of row m0+lr at K offset 16*half.
  const __bf16* bbase = Vswz + ((size_t)cb * NKC * 32 + lane) * 16;
  const __bf16* xrow  = xb_t + (size_t)(m0 + lr) * IN_SZ + 16 * half;
  const __bf16* hrow  = h_in + (size_t)(m0 + lr) * HS    + 16 * half;

  v8f acc = {};

  // Phase 1: x_t @ W   (K = 0..255)
  #pragma unroll
  for (int kc = 0; kc < IN_SZ / 32; ++kc) {
    Frag16 a, b;
    const __bf16* xp = xrow + kc * 32;
    a.u4[0] = *(const v4u*)(xp);
    a.u4[1] = *(const v4u*)(xp + 8);
    const __bf16* bp = bbase + (size_t)kc * CHUNK;
    b.u4[0] = *(const v4u*)(bp);
    b.u4[1] = *(const v4u*)(bp + 8);
    __builtin_prefetch(bp + CHUNK, 0, 1);   // global_prefetch_b8: next B frag
    acc = __builtin_amdgcn_wmma_f32_16x16x32_bf16(
        false, a.bf, false, b.bf, (short)0, acc, false, false);
  }

  // Phase 2: h @ U   (K = 256..767)
  #pragma unroll
  for (int kc = 0; kc < HS / 32; ++kc) {
    Frag16 a, b;
    const __bf16* hp = hrow + kc * 32;
    a.u4[0] = *(const v4u*)(hp);
    a.u4[1] = *(const v4u*)(hp + 8);
    const __bf16* bp = bbase + (size_t)(IN_SZ / 32 + kc) * CHUNK;
    b.u4[0] = *(const v4u*)(bp);
    b.u4[1] = *(const v4u*)(bp + 8);
    if (kc + 1 < HS / 32)
      __builtin_prefetch(bp + CHUNK, 0, 1);
    acc = __builtin_amdgcn_wmma_f32_16x16x32_bf16(
        false, a.bf, false, b.bf, (short)0, acc, false, false);
  }

  // Bias + gate nonlinearity (reference scales pre-activations by 2).
  {
    const float bb = bsum[wave * HS + n0 + lr];
    #pragma unroll
    for (int j = 0; j < 8; ++j) {
      int ml = j + 8 * half;              // C/D layout: M = vgpr + 8*(lane/16)
      float v = acc[j] + bb;
      float act = (wave == 2) ? tanhf(2.f * v)
                              : 1.f / (1.f + __expf(-2.f * v));
      Glds[wave][ml][lr] = act;
    }
  }
  __syncthreads();

  // Cell update: 256 outputs per block, 2 per thread.
  #pragma unroll
  for (int e = 0; e < 2; ++e) {
    int idx = e * 128 + tid;
    int m = idx >> 4, n = idx & 15;
    size_t off = (size_t)(m0 + m) * HS + (n0 + n);
    float iv = Glds[0][m][n], fv = Glds[1][m][n];
    float gv = Glds[2][m][n], ov = Glds[3][m][n];
    float cfull = fv * c_state[off] + iv * gv;
    float h = ov * tanhf(cfull);
    h_out_f32[off] = h;
    h_out_bf[off]  = (__bf16)h;
    c_state[off]   = 0.5f * cfull;        // reference carries c/2
  }
}

extern "C" void kernel_launch(void* const* d_in, const int* in_sizes, int n_in,
                              void* d_out, int out_size, void* d_ws, size_t ws_size,
                              hipStream_t stream) {
  (void)in_sizes; (void)n_in; (void)out_size; (void)ws_size;
  const float* x  = (const float*)d_in[0];   // [SEQ, BS, IN_SZ]
  const float* W  = (const float*)d_in[1];   // [IN_SZ, 4H]
  const float* U  = (const float*)d_in[2];   // [HS, 4H]
  const float* bw = (const float*)d_in[3];   // [4H]
  const float* bu = (const float*)d_in[4];   // [4H]
  float* out = (float*)d_out;                // hidden_seq | h_last | c_last

  // Workspace: Vswz bf16 (3 MB) | x bf16 (33.5 MB) | h x2 bf16 | c f32 | bias f32
  char* ws = (char*)d_ws;
  __bf16* Vswz = (__bf16*)ws;
  __bf16* xbf  = Vswz + (size_t)KTOT * FOURH;
  __bf16* hb0  = xbf + (size_t)SEQ * BS * IN_SZ;
  __bf16* hb1  = hb0 + (size_t)BS * HS;
  float*  cbuf = (float*)(hb1 + (size_t)BS * HS);
  float*  bsum = cbuf + (size_t)BS * HS;

  build_weights<<<(KTOT * FOURH + 255) / 256, 256, 0, stream>>>(W, U, Vswz);
  convert_x<<<(SEQ * BS * IN_SZ / 4 + 255) / 256, 256, 0, stream>>>(x, xbf);
  init_state<<<(BS * HS + 255) / 256, 256, 0, stream>>>(cbuf, hb0, bw, bu, bsum);

  __bf16* hb[2] = {hb0, hb1};
  dim3 grid(HS / 16, BS / 16);   // 32 x 8 workgroups, 128 threads each
  for (int t = 0; t < SEQ; ++t) {
    lstm_step<<<grid, 128, 0, stream>>>(
        xbf + (size_t)t * BS * IN_SZ, Vswz, bsum,
        hb[t & 1], hb[(t + 1) & 1], cbuf,
        out + (size_t)t * BS * HS);
  }

  // h_last = hidden_seq[-1]; c_last = carried cell state.
  float* hl = out + (size_t)SEQ * BS * HS;
  float* cl = hl + (size_t)BS * HS;
  hipMemcpyAsync(hl, out + (size_t)(SEQ - 1) * BS * HS,
                 (size_t)BS * HS * sizeof(float), hipMemcpyDeviceToDevice, stream);
  hipMemcpyAsync(cl, cbuf,
                 (size_t)BS * HS * sizeof(float), hipMemcpyDeviceToDevice, stream);
}
